// Attention_69260642615308
// MI455X (gfx1250) — compile-verified, matrix operand order checked by reading, not run on hardware
//
#include <hip/hip_runtime.h>
#include <hip/hip_bf16.h>
#include <stdint.h>

typedef __bf16 bf16;
typedef __attribute__((ext_vector_type(16))) __bf16 v16bf;
typedef __attribute__((ext_vector_type(8)))  __bf16 v8bf;
typedef __attribute__((ext_vector_type(8)))  float  v8f;

#define B_  4
#define T_  2048
#define D_  2048
#define H_  16
#define HD_ 128

// ---- CDNA5 async global->LDS copy (ASYNCcnt-tracked) -----------------------
// Per-lane: copies 16 bytes from the lane's global address to the lane's LDS
// address (low 32 bits of the generic shared pointer = workgroup LDS offset).
static __device__ __forceinline__ void async_copy_b128(const bf16* g, bf16* l) {
  unsigned lds = (unsigned)(uintptr_t)l;
  unsigned long long ga = (unsigned long long)(uintptr_t)g;
  asm volatile("global_load_async_to_lds_b128 %0, %1, off"
               :: "v"(lds), "v"(ga) : "memory");
}
#define WAIT_ASYNCCNT(n) asm volatile("s_wait_asynccnt " #n ::: "memory")

// ---- fragment helpers -------------------------------------------------------
// A/B fragment (16x32 / 32x16 bf16): lane L -> m/n = L&15, hi = L>>4.
// halves 0..7 = K (hi*8 + 0..7), halves 8..15 = K (16 + hi*8 + 0..7).
static __device__ __forceinline__ v16bf ld_frag16(const bf16* p0, const bf16* p1) {
  v8bf a = *(const v8bf*)p0;
  v8bf b = *(const v8bf*)p1;
  v16bf r;
#pragma unroll
  for (int i = 0; i < 8; ++i) { r[i] = a[i]; r[i + 8] = b[i]; }
  return r;
}

static __device__ __forceinline__ v8f wmma_bf16(v16bf a, v16bf b, v8f c) {
  return __builtin_amdgcn_wmma_f32_16x16x32_bf16(false, a, false, b, (short)0, c,
                                                 false, false);
}

// ============================================================================
// Kernel 0: fp32 -> bf16 bulk convert (one-time; makes x/W L2-resident bf16).
// ============================================================================
__global__ __launch_bounds__(256) void cvt_f32_bf16(const float* __restrict__ s,
                                                    bf16* __restrict__ d, int n) {
  int i = (blockIdx.x * 256 + threadIdx.x) * 8;
  if (i >= n) return;
  float4 a = *(const float4*)(s + i);
  float4 b = *(const float4*)(s + i + 4);
  v8bf o;
  o[0] = (bf16)a.x; o[1] = (bf16)a.y; o[2] = (bf16)a.z; o[3] = (bf16)a.w;
  o[4] = (bf16)b.x; o[5] = (bf16)b.y; o[6] = (bf16)b.z; o[7] = (bf16)b.w;
  *(v8bf*)(d + i) = o;
}

// ============================================================================
// Kernel A: C = x @ W^T (bf16 WMMA, fp32 accum), async double-buffered stage.
//   z==0: RoPE(double-angle) epilogue -> qk (B,H,T,HD)  [Q == K in reference]
//   z==1: plain epilogue, transposed  -> vt (B,H,HD,T)
// ============================================================================
#define LDSA 40   // 32 + 8 pad halves; 80B rows keep 16B alignment

__global__ __launch_bounds__(256) void qv_gemm_rope(
    const bf16* __restrict__ xb, const bf16* __restrict__ wqb,
    const bf16* __restrict__ wvb, const float* __restrict__ fcos,
    const float* __restrict__ fsin, bf16* __restrict__ qk,
    bf16* __restrict__ vt) {
  __shared__ bf16 lA[2][128 * LDSA];
  __shared__ bf16 lB[2][128 * LDSA];

  const int doV  = blockIdx.z;
  const bf16* w  = doV ? wvb : wqb;
  const int row0 = blockIdx.y * 128;
  const int col0 = blockIdx.x * 128;
  const int tid  = threadIdx.x;
  const int lane = tid & 31, wid = tid >> 5;
  const int wm = wid & 3, wn = wid >> 2;      // wave tile: 32 rows x 64 cols
  const int lm = lane & 15, hi = lane >> 4;

  v8f acc[2][4];
#pragma unroll
  for (int i = 0; i < 2; ++i)
#pragma unroll
    for (int j = 0; j < 4; ++j) acc[i][j] = (v8f){0.f,0.f,0.f,0.f,0.f,0.f,0.f,0.f};

  const int rld = tid >> 1;                  // 128 rows, 2 threads per row
  const int cld = (tid & 1) * 16;            // 16 halves each
  const bf16* ga = xb + (size_t)(row0 + rld) * D_ + cld;
  const bf16* gb = w  + (size_t)(col0 + rld) * D_ + cld;

  auto issue = [&](int buf, int k0) {        // 4 async ops / thread / stage
    async_copy_b128(ga + k0,     &lA[buf][rld * LDSA + cld]);
    async_copy_b128(ga + k0 + 8, &lA[buf][rld * LDSA + cld + 8]);
    async_copy_b128(gb + k0,     &lB[buf][rld * LDSA + cld]);
    async_copy_b128(gb + k0 + 8, &lB[buf][rld * LDSA + cld + 8]);
  };

  issue(0, 0);
  issue(1, 32);
  for (int kk = 0; kk < 64; ++kk) {
    if (kk < 63) { WAIT_ASYNCCNT(4); } else { WAIT_ASYNCCNT(0); }
    __syncthreads();
    const int buf = kk & 1;

    v16bf af[2], bfr[4];
#pragma unroll
    for (int i = 0; i < 2; ++i) {
      const bf16* p = &lA[buf][(wm * 32 + i * 16 + lm) * LDSA];
      af[i] = ld_frag16(p + hi * 8, p + 16 + hi * 8);
    }
#pragma unroll
    for (int j = 0; j < 4; ++j) {
      const bf16* p = &lB[buf][(wn * 64 + j * 16 + lm) * LDSA];
      bfr[j] = ld_frag16(p + hi * 8, p + 16 + hi * 8);
    }
#pragma unroll
    for (int i = 0; i < 2; ++i)
#pragma unroll
      for (int j = 0; j < 4; ++j) acc[i][j] = wmma_bf16(af[i], bfr[j], acc[i][j]);
    __syncthreads();
    if (kk + 2 < 64) issue(buf, (kk + 2) * 32);
  }

  // epilogue: C layout = VGPR r -> row (r + 8*hi), lane&15 -> col
#pragma unroll
  for (int i = 0; i < 2; ++i) {
#pragma unroll
    for (int j = 0; j < 4; ++j) {
      const int n = col0 + wn * 64 + j * 16 + lm;     // feature index in D
      const int h = n >> 7, hd = n & 127;
#pragma unroll
      for (int r = 0; r < 8; ++r) {
        const int gm = row0 + wm * 32 + i * 16 + r + hi * 8;  // flat (b,t)
        const int b = gm >> 11, t = gm & (T_ - 1);
        float val = acc[i][j][r];
        if (doV) {
          vt[(((size_t)(b * H_ + h)) * HD_ + hd) * T_ + t] = (bf16)val;
        } else {
          float part = __shfl_xor(val, 1);  // RoPE pair lives in adjacent lane
          const int jf = hd >> 1;
          float cc = fcos[t * (HD_ / 2) + jf];
          float ss = fsin[t * (HD_ / 2) + jf];
          float c2 = cc * cc - ss * ss;     // double-angle per reference
          float s2 = 2.f * cc * ss;
          float res = (n & 1) ? (part * s2 + val * c2)
                              : (val * c2 - part * s2);
          qk[(((size_t)(b * H_ + h)) * T_ + t) * HD_ + hd] = (bf16)res;
        }
      }
    }
  }
}

// ============================================================================
// Kernel B: causal flash attention, Q == K == qk, V = vt (transposed),
// async double-buffered K/V tile staging.
// ============================================================================
__global__ __launch_bounds__(256) void flash_attn(
    const bf16* __restrict__ qk, const bf16* __restrict__ vt,
    bf16* __restrict__ attn) {
  __shared__ bf16 lK[2][32 * 136];   // 32 kv rows x 128 halves (+8 pad)
  __shared__ bf16 lV[2][128 * 40];   // 128 d rows x 32 kv halves (+8 pad)
  __shared__ bf16 lP[8][16 * 32];    // per-wave P scratch (C-layout -> A-frag)

  const int b = blockIdx.z, h = blockIdx.y;
  const int q0 = blockIdx.x * 128;
  const int tid = threadIdx.x, lane = tid & 31, wid = tid >> 5;
  const int lm = lane & 15, hi = lane >> 4;
  const int qrow = q0 + wid * 16;

  const bf16* Q = qk + ((size_t)(b * H_ + h) * T_) * HD_;
  const bf16* V = vt + ((size_t)(b * H_ + h) * HD_) * T_;

  // preload Q fragments (16 rows x 128 d)
  v16bf qf[4];
#pragma unroll
  for (int kc = 0; kc < 4; ++kc) {
    const bf16* p = Q + (size_t)(qrow + lm) * HD_ + kc * 32;
    qf[kc] = ld_frag16(p + hi * 8, p + 16 + hi * 8);
  }

  v8f o[8];
#pragma unroll
  for (int d = 0; d < 8; ++d) o[d] = (v8f){0.f,0.f,0.f,0.f,0.f,0.f,0.f,0.f};
  float ms[8], ls[8];
#pragma unroll
  for (int r = 0; r < 8; ++r) { ms[r] = -1e30f; ls[r] = 0.f; }

  const float scale = 0.08838834764831845f;  // 1/sqrt(128)
  const int nst = (q0 + 128) / 32;           // causal stage count (>= 4)

  const int rk = tid >> 3, ck = (tid & 7) * 16;   // K tile: 32 x 128
  const int rv = tid >> 1, cv = (tid & 1) * 16;   // V tile: 128 x 32

  auto issue = [&](int buf, int st) {        // 4 async ops / thread / stage
    const int k0 = st * 32;
    async_copy_b128(Q + (size_t)(k0 + rk) * HD_ + ck,     &lK[buf][rk * 136 + ck]);
    async_copy_b128(Q + (size_t)(k0 + rk) * HD_ + ck + 8, &lK[buf][rk * 136 + ck + 8]);
    async_copy_b128(V + (size_t)rv * T_ + k0 + cv,        &lV[buf][rv * 40 + cv]);
    async_copy_b128(V + (size_t)rv * T_ + k0 + cv + 8,    &lV[buf][rv * 40 + cv + 8]);
  };

  issue(0, 0);
  if (nst > 1) issue(1, 1);
  for (int st = 0; st < nst; ++st) {
    if (st + 1 < nst) { WAIT_ASYNCCNT(4); } else { WAIT_ASYNCCNT(0); }
    __syncthreads();
    const int buf = st & 1;
    const int k0 = st * 32;

    if (k0 <= qrow + 15) {                  // wave-uniform causal skip
      // S = Q K^T  (16 x 32)
      v8f s[2];
      s[0] = (v8f){0.f,0.f,0.f,0.f,0.f,0.f,0.f,0.f};
      s[1] = s[0];
#pragma unroll
      for (int nt = 0; nt < 2; ++nt)
#pragma unroll
        for (int kc = 0; kc < 4; ++kc) {
          const bf16* p = &lK[buf][(nt * 16 + lm) * 136 + kc * 32];
          v16bf kf = ld_frag16(p + hi * 8, p + 16 + hi * 8);
          s[nt] = wmma_bf16(qf[kc], kf, s[nt]);
        }

      // scale + causal mask + online softmax
      float p0v[8], p1v[8];
#pragma unroll
      for (int r = 0; r < 8; ++r) {
        const int q = qrow + r + hi * 8;
        float s0 = s[0][r] * scale + ((k0 + lm)      > q ? -1e9f : 0.f);
        float s1 = s[1][r] * scale + ((k0 + 16 + lm) > q ? -1e9f : 0.f);
        float mx = fmaxf(s0, s1);
#pragma unroll
        for (int m = 1; m < 16; m <<= 1) mx = fmaxf(mx, __shfl_xor(mx, m));
        float mnew  = fmaxf(ms[r], mx);
        float alpha = __expf(ms[r] - mnew);
        float p0 = __expf(s0 - mnew);
        float p1 = __expf(s1 - mnew);
        float rs = p0 + p1;
#pragma unroll
        for (int m = 1; m < 16; m <<= 1) rs += __shfl_xor(rs, m);
        ls[r] = ls[r] * alpha + rs;
        ms[r] = mnew;
#pragma unroll
        for (int d = 0; d < 8; ++d) o[d][r] = o[d][r] * alpha;
        p0v[r] = p0; p1v[r] = p1;
      }
      // spill P (C layout) to per-wave LDS, reload as A fragment
#pragma unroll
      for (int r = 0; r < 8; ++r) {
        const int prow = r + hi * 8;
        lP[wid][prow * 32 + lm]      = (bf16)p0v[r];
        lP[wid][prow * 32 + 16 + lm] = (bf16)p1v[r];
      }
      v16bf pf = ld_frag16(&lP[wid][lm * 32 + hi * 8],
                           &lP[wid][lm * 32 + 16 + hi * 8]);
      // O += P * V
#pragma unroll
      for (int dt = 0; dt < 8; ++dt) {
        const bf16* p = &lV[buf][(dt * 16 + lm) * 40];
        v16bf vf = ld_frag16(p + hi * 8, p + 16 + hi * 8);
        o[dt] = wmma_bf16(pf, vf, o[dt]);
      }
    }
    __syncthreads();
    if (st + 2 < nst) issue(buf, st + 2);
  }

  // write attention output, (B*T, D) bf16, normalized by l
#pragma unroll
  for (int dt = 0; dt < 8; ++dt) {
    const int d = h * HD_ + dt * 16 + lm;
#pragma unroll
    for (int r = 0; r < 8; ++r) {
      const int t = qrow + r + hi * 8;
      float val = o[dt][r] / ls[r];
      attn[((size_t)b * T_ + t) * D_ + d] = (bf16)val;
    }
  }
}

// ============================================================================
// Kernel C: d_out = attn(bf16) @ wo^T (bf16), fp32 output; async staging.
// ============================================================================
__global__ __launch_bounds__(256) void out_gemm(
    const bf16* __restrict__ attn, const bf16* __restrict__ wob,
    float* __restrict__ out) {
  __shared__ bf16 lA[2][128 * LDSA];
  __shared__ bf16 lB[2][128 * LDSA];

  const int row0 = blockIdx.y * 128;
  const int col0 = blockIdx.x * 128;
  const int tid  = threadIdx.x;
  const int lane = tid & 31, wid = tid >> 5;
  const int wm = wid & 3, wn = wid >> 2;
  const int lm = lane & 15, hi = lane >> 4;

  v8f acc[2][4];
#pragma unroll
  for (int i = 0; i < 2; ++i)
#pragma unroll
    for (int j = 0; j < 4; ++j) acc[i][j] = (v8f){0.f,0.f,0.f,0.f,0.f,0.f,0.f,0.f};

  const int rld = tid >> 1;
  const int cld = (tid & 1) * 16;
  const bf16* ga = attn + (size_t)(row0 + rld) * D_ + cld;
  const bf16* gb = wob  + (size_t)(col0 + rld) * D_ + cld;

  auto issue = [&](int buf, int k0) {
    async_copy_b128(ga + k0,     &lA[buf][rld * LDSA + cld]);
    async_copy_b128(ga + k0 + 8, &lA[buf][rld * LDSA + cld + 8]);
    async_copy_b128(gb + k0,     &lB[buf][rld * LDSA + cld]);
    async_copy_b128(gb + k0 + 8, &lB[buf][rld * LDSA + cld + 8]);
  };

  issue(0, 0);
  issue(1, 32);
  for (int kk = 0; kk < 64; ++kk) {
    if (kk < 63) { WAIT_ASYNCCNT(4); } else { WAIT_ASYNCCNT(0); }
    __syncthreads();
    const int buf = kk & 1;

    v16bf af[2], bfr[4];
#pragma unroll
    for (int i = 0; i < 2; ++i) {
      const bf16* p = &lA[buf][(wm * 32 + i * 16 + lm) * LDSA];
      af[i] = ld_frag16(p + hi * 8, p + 16 + hi * 8);
    }
#pragma unroll
    for (int j = 0; j < 4; ++j) {
      const bf16* p = &lB[buf][(wn * 64 + j * 16 + lm) * LDSA];
      bfr[j] = ld_frag16(p + hi * 8, p + 16 + hi * 8);
    }
#pragma unroll
    for (int i = 0; i < 2; ++i)
#pragma unroll
      for (int j = 0; j < 4; ++j) acc[i][j] = wmma_bf16(af[i], bfr[j], acc[i][j]);
    __syncthreads();
    if (kk + 2 < 64) issue(buf, (kk + 2) * 32);
  }

#pragma unroll
  for (int i = 0; i < 2; ++i)
#pragma unroll
    for (int j = 0; j < 4; ++j) {
      const int n = col0 + wn * 64 + j * 16 + lm;
#pragma unroll
      for (int r = 0; r < 8; ++r) {
        const int gm = row0 + wm * 32 + i * 16 + r + hi * 8;
        out[(size_t)gm * D_ + n] = acc[i][j][r];
      }
    }
}

// ============================================================================
extern "C" void kernel_launch(void* const* d_in, const int* in_sizes, int n_in,
                              void* d_out, int out_size, void* d_ws, size_t ws_size,
                              hipStream_t stream) {
  const float* x  = (const float*)d_in[0];
  const float* wq = (const float*)d_in[1];
  // d_in[2] = wk: dead in the reference (xk is overwritten with rotated xq)
  const float* wv = (const float*)d_in[3];
  const float* wo = (const float*)d_in[4];
  const float* fc = (const float*)d_in[5];
  const float* fs = (const float*)d_in[6];
  // mask / caches / start_pos: start_pos==0 and caches start zero, so
  // keys == rot(Q), values == V; causal mask applied analytically.

  char* ws = (char*)d_ws;
  const size_t nx = (size_t)B_ * T_ * D_;      // 16.78M elements
  const size_t nw = (size_t)D_ * D_;           // 4.19M elements
  bf16* xb   = (bf16*)(ws);
  bf16* wqb  = (bf16*)(ws + nx * 2);
  bf16* wvb  = (bf16*)(ws + nx * 2 + nw * 2);
  bf16* wob  = (bf16*)(ws + nx * 2 + nw * 4);
  bf16* qkb  = (bf16*)(ws + nx * 2 + nw * 6);
  bf16* vtb  = (bf16*)(ws + nx * 2 + nw * 6 + nx * 2);
  bf16* attn = (bf16*)(ws + nx * 2 + nw * 6 + nx * 4);
  float* out = (float*)d_out;

  cvt_f32_bf16<<<(unsigned)(nx / 8 / 256), 256, 0, stream>>>(x,  xb,  (int)nx);
  cvt_f32_bf16<<<(unsigned)(nw / 8 / 256), 256, 0, stream>>>(wq, wqb, (int)nw);
  cvt_f32_bf16<<<(unsigned)(nw / 8 / 256), 256, 0, stream>>>(wv, wvb, (int)nw);
  cvt_f32_bf16<<<(unsigned)(nw / 8 / 256), 256, 0, stream>>>(wo, wob, (int)nw);

  dim3 gA(D_ / 128, (B_ * T_) / 128, 2);
  qv_gemm_rope<<<gA, 256, 0, stream>>>(xb, wqb, wvb, fc, fs, qkb, vtb);

  dim3 gB(T_ / 128, H_, B_);
  flash_attn<<<gB, 256, 0, stream>>>(qkb, vtb, attn);

  dim3 gC(D_ / 128, (B_ * T_) / 128);
  out_gemm<<<gC, 256, 0, stream>>>(attn, wob, out);
}
